// SuzukiYieldGNN_4999341932789
// MI455X (gfx1250) — matrix operand order repaired and checked by reading, not compiled, not run on hardware
//
#include <hip/hip_runtime.h>
#include <hip/hip_bf16.h>
#include <stdint.h>

// ---------------------------------------------------------------------------
// SuzukiYieldGNN forward for MI455X (gfx1250, wave32, WMMA).
//
// Heavy GEMMs use v_wmma_f32_16x16x32_bf16 with f32 accumulation.
//  - Weights are pre-swizzled once into WMMA-fragment-native layout so the
//    inner loop fetches each B fragment with two global_load_b128 per lane.
//  - A tiles are swizzled while filling LDS so each A fragment is two
//    ds_load_b128 per lane.
//  - The 34-GFLOP/layer edge GEMM is fused with the GAT logit epilogue and
//    never materializes the [65000,1024] tensor; its xl/xr gathers are
//    served from the 192MB L2 (each operand is 20.5MB).
// ---------------------------------------------------------------------------

typedef __attribute__((ext_vector_type(16))) __bf16 v16bf;
typedef __attribute__((ext_vector_type(8)))  float  v8f;

#define DEV __device__ __forceinline__

DEV unsigned short f2bf(float f) {             // round-to-nearest-even f32->bf16
  unsigned u = __float_as_uint(f);
  unsigned r = u + 0x7FFFu + ((u >> 16) & 1u);
  return (unsigned short)(r >> 16);
}

// order-preserving float<->uint mapping for atomic max on floats
DEV unsigned f2ord(float f) {
  unsigned u = __float_as_uint(f);
  return (u & 0x80000000u) ? ~u : (u | 0x80000000u);
}
DEV float ord2f(unsigned u) {
  return (u & 0x80000000u) ? __uint_as_float(u & 0x7FFFFFFFu) : __uint_as_float(~u);
}
#define NEG_INF_ORD 0x007FFFFFu   // f2ord(-inf)

union AFrag { unsigned u[8]; uint4 q[2]; v16bf v; };  // 16 bf16 per lane
union CFrag { float f[8];    v8f  v; };               // 8 f32 per lane

DEV v8f wmma_bf16(v16bf a, v16bf b, v8f c) {
  return __builtin_amdgcn_wmma_f32_16x16x32_bf16(false, a, false, b, (short)0, c,
                                                 false, false);
}

// ---------------------------------------------------------------------------
// elementwise helpers
// ---------------------------------------------------------------------------
__global__ void k_f32_to_bf16(const float* __restrict__ s,
                              unsigned short* __restrict__ d, long long n) {
  long long i = (long long)blockIdx.x * blockDim.x + threadIdx.x;
  if (i < n) d[i] = f2bf(s[i]);
}

__global__ void k_fill_u32(unsigned* p, unsigned v, long long n) {
  long long i = (long long)blockIdx.x * blockDim.x + threadIdx.x;
  if (i < n) p[i] = v;
}

// ---------------------------------------------------------------------------
// Weight swizzle: f32 W[K,N] -> bf16 fragment-native blocks.
// Block (nt, ksb) covers cols nt*16..+15, rows ksb*32..+31, stored as
// 32 lanes x 8 dwords: lane (hf*16 + n), dword j holds rows ((j>>2)*16+hf*8+2*(j&3), +1).
// Block order: nt-major, ksb-minor (sequential k streaming per column tile).
// ---------------------------------------------------------------------------
__global__ void k_swizzle_w(const float* __restrict__ W, unsigned* __restrict__ out,
                            int K, int N) {
  long long i = (long long)blockIdx.x * blockDim.x + threadIdx.x;
  long long total = (long long)K * N / 2;
  if (i >= total) return;
  int j = (int)(i & 7);
  int lane = (int)((i >> 3) & 31);
  long long blk = i >> 8;
  int KB = K >> 5;
  int nt = (int)(blk / KB), ksb = (int)(blk % KB);
  int n = nt * 16 + (lane & 15);
  int k = ksb * 32 + (j >> 2) * 16 + (lane >> 4) * 8 + (j & 3) * 2;
  unsigned lo = f2bf(W[(size_t)k * N + n]);
  unsigned hi = f2bf(W[(size_t)(k + 1) * N + n]);
  out[i] = lo | (hi << 16);
}

// write one dword of a row-major bf16 A row into swizzled LDS position
DEV int a_swz_index(int ml /*0..63*/, int kp /*dword pair idx*/) {
  int k = kp * 2;
  int ksb = k >> 5, kk = k & 31;
  int lane = ((kk & 15) >> 3) * 16 + (ml & 15);
  int j = (kk >> 4) * 4 + ((kk & 7) >> 1);
  return ((ksb * 4 + (ml >> 4)) << 8) + lane * 8 + j;
}

// ---------------------------------------------------------------------------
// WMMA GEMM: C[M,N] = A_bf16[M,K] @ Bswz[K,N] + bias[N]
// block = 128 threads (4 waves), tile 64(M) x 64(N), K in {64,256}
// ---------------------------------------------------------------------------
__global__ __launch_bounds__(128) void k_wmma_gemm_bias(
    const unsigned short* __restrict__ A, const unsigned* __restrict__ Bswz,
    const float* __restrict__ bias, float* __restrict__ C, int M, int N, int K) {
  __shared__ __align__(16) unsigned sA[64 * 128];  // swizzled, K<=256
  const int m0 = blockIdx.x * 64;
  const int n0 = blockIdx.y * 64;
  const int tid  = threadIdx.x;
  const int wave = tid >> 5;
  const int lane = tid & 31;
  const int hf   = lane >> 4;
  const int l16  = lane & 15;
  const int Ku   = K >> 1;
  const int KB   = K >> 5;

  {
    const unsigned* Au = (const unsigned*)A;
    const int nu = 64 * Ku;
    for (int i = tid; i < nu; i += 128) {
      int ml = i / Ku, kp = i % Ku;
      int m = m0 + ml;
      unsigned v = (m < M) ? Au[(size_t)m * Ku + kp] : 0u;
      sA[a_swz_index(ml, kp)] = v;
    }
  }
  __syncthreads();

  CFrag acc[4];
#pragma unroll
  for (int t = 0; t < 4; t++)
#pragma unroll
    for (int r = 0; r < 8; r++) acc[t].f[r] = 0.0f;

  const uint4* sA4 = (const uint4*)sA;
  const uint4* B4  = (const uint4*)Bswz;
  const long long ntg = (long long)(blockIdx.y * 4 + wave);

  for (int ksb = 0; ksb < KB; ksb++) {
    AFrag bfr;
    long long bb = (ntg * KB + ksb) * 64 + lane * 2;
    bfr.q[0] = B4[bb];
    bfr.q[1] = B4[bb + 1];
#pragma unroll
    for (int mt = 0; mt < 4; mt++) {
      AFrag afr;
      int ab = (ksb * 4 + mt) * 64 + lane * 2;
      afr.q[0] = sA4[ab];
      afr.q[1] = sA4[ab + 1];
      acc[mt].v = wmma_bf16(afr.v, bfr.v, acc[mt].v);
    }
  }

  const int col = n0 + wave * 16 + l16;
  float bv = bias ? bias[col] : 0.0f;
  if (m0 + 64 <= M) {                         // fast path: tile fully in-bounds
#pragma unroll
    for (int mt = 0; mt < 4; mt++)
#pragma unroll
      for (int r = 0; r < 8; r++) {
        int m = m0 + mt * 16 + r + hf * 8;
        C[(size_t)m * N + col] = acc[mt].f[r] + bv;
      }
  } else {
#pragma unroll
    for (int mt = 0; mt < 4; mt++)
#pragma unroll
      for (int r = 0; r < 8; r++) {
        int m = m0 + mt * 16 + r + hf * 8;
        if (m < M) C[(size_t)m * N + col] = acc[mt].f[r] + bv;
      }
  }
}

// ---------------------------------------------------------------------------
// Fused GAT edge kernel: for each edge e, head h:
//   logit[e,h] = sum_c att[h,c]*lrelu( (e_f@we)[e,h*256+c] + xl[src,..] + xr[dst,..] )
// block = 64 edges x 4 heads (one wave per head). K=256, 16 N-tiles per head.
// ---------------------------------------------------------------------------
__global__ __launch_bounds__(128) void k_gat_edge_logits(
    const unsigned short* __restrict__ Ef_bf,   // [Ef,256] bf16 row-major
    const unsigned* __restrict__ We_swz,        // swizzled [256,1024]
    const float* __restrict__ xl, const float* __restrict__ xr,  // [N,1024]
    const int* __restrict__ src_f, const int* __restrict__ dst_f,
    const float* __restrict__ att,              // [4,256] flat
    float* __restrict__ logit,                  // [Ef,4]
    int Ef) {
  __shared__ __align__(16) unsigned sA[64 * 128];
  __shared__ int sSrc[64], sDst[64];
  const int e0   = blockIdx.x * 64;
  const int tid  = threadIdx.x;
  const int h    = tid >> 5;                    // wave == head
  const int lane = tid & 31;
  const int hf   = lane >> 4;
  const int l16  = lane & 15;

  {
    const unsigned* Au = (const unsigned*)Ef_bf;
    for (int i = tid; i < 64 * 128; i += 128) {
      int ml = i >> 7, kp = i & 127;
      int e = e0 + ml;
      unsigned v = (e < Ef) ? Au[(size_t)e * 128 + kp] : 0u;
      sA[a_swz_index(ml, kp)] = v;
    }
    for (int i = tid; i < 64; i += 128) {
      int e = e0 + i;
      sSrc[i] = (e < Ef) ? src_f[e] : 0;
      sDst[i] = (e < Ef) ? dst_f[e] : 0;
    }
  }
  __syncthreads();

  float esum[4][8];
#pragma unroll
  for (int mt = 0; mt < 4; mt++)
#pragma unroll
    for (int r = 0; r < 8; r++) esum[mt][r] = 0.0f;

  const uint4* sA4 = (const uint4*)sA;
  const uint4* B4  = (const uint4*)We_swz;

  for (int nt = 0; nt < 16; nt++) {
    const int col = h * 256 + nt * 16 + l16;
    const long long ntg = (long long)(h * 16 + nt);
    CFrag acc[4];
#pragma unroll
    for (int t = 0; t < 4; t++)
#pragma unroll
      for (int r = 0; r < 8; r++) acc[t].f[r] = 0.0f;

    if (nt < 15)  // prefetch next column-tile's weight stream (global_prefetch)
      __builtin_prefetch((const void*)&B4[(ntg + 1) * 8 * 64 + lane * 2], 0, 1);

#pragma unroll
    for (int ksb = 0; ksb < 8; ksb++) {
      AFrag bfr;
      long long bb = (ntg * 8 + ksb) * 64 + lane * 2;
      bfr.q[0] = B4[bb];
      bfr.q[1] = B4[bb + 1];
#pragma unroll
      for (int mt = 0; mt < 4; mt++) {
        AFrag afr;
        int ab = (ksb * 4 + mt) * 64 + lane * 2;
        afr.q[0] = sA4[ab];
        afr.q[1] = sA4[ab + 1];
        acc[mt].v = wmma_bf16(afr.v, bfr.v, acc[mt].v);
      }
    }

    // epilogue: gather + leaky-relu + att-dot (reduce across the 16 n-lanes)
    float attv = att[col];
#pragma unroll
    for (int mt = 0; mt < 4; mt++)
#pragma unroll
      for (int r = 0; r < 8; r++) {
        int eloc = mt * 16 + r + hf * 8;
        int s = sSrc[eloc], d = sDst[eloc];
        float v = acc[mt].f[r] + xl[(size_t)s * 1024 + col] +
                  xr[(size_t)d * 1024 + col];
        v = v > 0.0f ? v : 0.2f * v;             // leaky_relu(0.2)
        float t = v * attv;
        t += __shfl_xor(t, 1, 32);
        t += __shfl_xor(t, 2, 32);
        t += __shfl_xor(t, 4, 32);
        t += __shfl_xor(t, 8, 32);
        esum[mt][r] += t;
      }
  }

  if (l16 == 0) {                                // lanes 0 & 16 own the sums
#pragma unroll
    for (int mt = 0; mt < 4; mt++)
#pragma unroll
      for (int r = 0; r < 8; r++) {
        int e = e0 + mt * 16 + r + hf * 8;
        if (e < Ef) logit[(size_t)e * 4 + h] = esum[mt][r];
      }
  }
}

// ---------------------------------------------------------------------------
// small f32 GEMM: C = A@B (+bias) (+=C if accum). one thread per output.
// ---------------------------------------------------------------------------
__global__ void k_sgemm(const float* __restrict__ A, const float* __restrict__ B,
                        const float* __restrict__ bias, float* __restrict__ C,
                        int M, int N, int K, int accum) {
  long long i = (long long)blockIdx.x * blockDim.x + threadIdx.x;
  if (i >= (long long)M * N) return;
  int m = (int)(i / N), n = (int)(i % N);
  float s = 0.0f;
  for (int k = 0; k < K; k++) s += A[(size_t)m * K + k] * B[(size_t)k * N + n];
  if (bias) s += bias[n];
  if (accum) s += C[i];
  C[i] = s;
}

// column mean/var (biased) over M rows; one block per column
__global__ void k_colstats(const float* __restrict__ X, float* mean, float* var,
                           int M, int C) {
  __shared__ float s1[256], s2[256];
  int c = blockIdx.x;
  float a = 0.f, b = 0.f;
  for (int m = threadIdx.x; m < M; m += blockDim.x) {
    float v = X[(size_t)m * C + c];
    a += v; b += v * v;
  }
  s1[threadIdx.x] = a; s2[threadIdx.x] = b;
  __syncthreads();
  for (int st = 128; st > 0; st >>= 1) {
    if ((int)threadIdx.x < st) {
      s1[threadIdx.x] += s1[threadIdx.x + st];
      s2[threadIdx.x] += s2[threadIdx.x + st];
    }
    __syncthreads();
  }
  if (threadIdx.x == 0) {
    float mu = s1[0] / M;
    mean[c] = mu;
    var[c] = s2[0] / M - mu * mu;
  }
}

// y = act((x-mean)*rsqrt(var+eps)*g+b) + res   act: 0 none, 1 relu, 2 elu
__global__ void k_bn_apply(const float* __restrict__ X, const float* mean,
                           const float* var, const float* g, const float* b,
                           const float* __restrict__ res, float* __restrict__ Y,
                           int M, int C, int act) {
  long long i = (long long)blockIdx.x * blockDim.x + threadIdx.x;
  if (i >= (long long)M * C) return;
  int c = (int)(i % C);
  float v = (X[i] - mean[c]) * rsqrtf(var[c] + 1e-5f) * g[c] + b[c];
  if (act == 1) v = fmaxf(v, 0.0f);
  else if (act == 2) v = v > 0.0f ? v : (expf(v) - 1.0f);
  if (res) v += res[i];
  Y[i] = v;
}

// ---------------------------------------------------------------------------
// graph plumbing
// ---------------------------------------------------------------------------
__global__ void k_build_edges(const int* __restrict__ ei, int* src_f, int* dst_f,
                              int E, int N) {
  int i = blockIdx.x * blockDim.x + threadIdx.x;
  if (i >= E + N) return;
  if (i < E) { src_f[i] = ei[i]; dst_f[i] = ei[E + i]; }
  else       { src_f[i] = i - E; dst_f[i] = i - E; }
}

__global__ void k_deg(const int* __restrict__ dst, float* deg, int E) {
  int i = blockIdx.x * blockDim.x + threadIdx.x;
  if (i < E) atomicAdd(&deg[dst[i]], 1.0f);
}

__global__ void k_loop_sum(const float* __restrict__ e, const int* __restrict__ dst,
                           float* __restrict__ loop, int E) {
  long long i = (long long)blockIdx.x * blockDim.x + threadIdx.x;
  if (i >= (long long)E * 256) return;
  int ed = (int)(i >> 8), c = (int)(i & 255);
  atomicAdd(&loop[(size_t)dst[ed] * 256 + c], e[(size_t)ed * 256 + c]);
}

__global__ void k_loop_div(float* loop, const float* deg, int N) {
  long long i = (long long)blockIdx.x * blockDim.x + threadIdx.x;
  if (i >= (long long)N * 256) return;
  loop[i] /= fmaxf(deg[i >> 8], 1.0f);
}

__global__ void k_edge_seg_max(const float* __restrict__ logit,
                               const int* __restrict__ dst_f, unsigned* segm, int Ef) {
  long long i = (long long)blockIdx.x * blockDim.x + threadIdx.x;
  if (i >= (long long)Ef * 4) return;
  int e = (int)(i >> 2), h = (int)(i & 3);
  atomicMax(&segm[(size_t)dst_f[e] * 4 + h], f2ord(logit[i]));
}

__global__ void k_edge_exp(const float* __restrict__ logit,
                           const int* __restrict__ dst_f, const unsigned* segm,
                           float* ex, float* den, int Ef) {
  long long i = (long long)blockIdx.x * blockDim.x + threadIdx.x;
  if (i >= (long long)Ef * 4) return;
  int e = (int)(i >> 2), h = (int)(i & 3);
  float m = ord2f(segm[(size_t)dst_f[e] * 4 + h]);
  float v = expf(logit[i] - m);
  ex[i] = v;
  atomicAdd(&den[(size_t)dst_f[e] * 4 + h], v);
}

__global__ void k_gat_scatter(const float* __restrict__ xl,
                              const int* __restrict__ src_f,
                              const int* __restrict__ dst_f,
                              const float* __restrict__ ex,
                              const float* __restrict__ den,
                              float* __restrict__ outp, long long total) {
  long long i = (long long)blockIdx.x * blockDim.x + threadIdx.x;
  if (i >= total) return;
  int e = (int)(i >> 10), j = (int)(i & 1023), h = j >> 8;
  float alpha = ex[(size_t)e * 4 + h] / den[(size_t)dst_f[e] * 4 + h];
  atomicAdd(&outp[(size_t)dst_f[e] * 1024 + j],
            xl[(size_t)src_f[e] * 1024 + j] * alpha);
}

__global__ void k_head_mean(const float* __restrict__ outp,
                            const float* __restrict__ bias, float* __restrict__ y,
                            int N) {
  long long i = (long long)blockIdx.x * blockDim.x + threadIdx.x;
  if (i >= (long long)N * 256) return;
  int n = (int)(i >> 8), c = (int)(i & 255);
  const float* o = outp + (size_t)n * 1024;
  y[i] = 0.25f * (o[c] + o[256 + c] + o[512 + c] + o[768 + c]) + bias[c];
}

// ---------------------------------------------------------------------------
// Set2Set
// ---------------------------------------------------------------------------
__global__ void k_lstm_update(const float* __restrict__ gates, float* hl, float* cl,
                              int G) {
  long long i = (long long)blockIdx.x * blockDim.x + threadIdx.x;
  if (i >= (long long)G * 256) return;
  int g = (int)(i >> 8), c = (int)(i & 255);
  const float* gr = gates + (size_t)g * 1024;
  float ig = 1.0f / (1.0f + expf(-gr[c]));
  float fg = 1.0f / (1.0f + expf(-gr[256 + c]));
  float gg = tanhf(gr[512 + c]);
  float og = 1.0f / (1.0f + expf(-gr[768 + c]));
  float cc = fg * cl[i] + ig * gg;
  cl[i] = cc;
  hl[i] = og * tanhf(cc);
}

__global__ void k_s2s_en(const float* __restrict__ hres, const float* __restrict__ hl,
                         const int* __restrict__ batch, float* en, int N) {
  int n = blockIdx.x * blockDim.x + threadIdx.x;
  if (n >= N) return;
  const float* hr = hres + (size_t)n * 256;
  const float* q = hl + (size_t)batch[n] * 256;
  float s = 0.0f;
  for (int c = 0; c < 256; c++) s += hr[c] * q[c];
  en[n] = s;
}

__global__ void k_seg_max_n(const float* en, const int* batch, unsigned* gm, int N) {
  int n = blockIdx.x * blockDim.x + threadIdx.x;
  if (n < N) atomicMax(&gm[batch[n]], f2ord(en[n]));
}

__global__ void k_seg_exp_n(const float* en, const int* batch, const unsigned* gm,
                            float* ea, float* gden, int N) {
  int n = blockIdx.x * blockDim.x + threadIdx.x;
  if (n >= N) return;
  float v = expf(en[n] - ord2f(gm[batch[n]]));
  ea[n] = v;
  atomicAdd(&gden[batch[n]], v);
}

__global__ void k_s2s_r(const float* __restrict__ hres, const int* __restrict__ batch,
                        const float* ea, const float* gden, float* r, int N) {
  long long i = (long long)blockIdx.x * blockDim.x + threadIdx.x;
  if (i >= (long long)N * 256) return;
  int n = (int)(i >> 8), c = (int)(i & 255);
  int g = batch[n];
  atomicAdd(&r[(size_t)g * 256 + c], hres[i] * ea[n] / gden[g]);
}

__global__ void k_qstar(const float* hl, const float* r, float* qs, int G) {
  long long i = (long long)blockIdx.x * blockDim.x + threadIdx.x;
  if (i >= (long long)G * 512) return;
  int g = (int)(i >> 9), c = (int)(i & 511);
  qs[i] = (c < 256) ? hl[(size_t)g * 256 + c] : r[(size_t)g * 256 + (c - 256)];
}

// per-row layernorm (in-place safe); block per row
__global__ void k_layernorm(const float* __restrict__ X, const float* g,
                            const float* b, float* __restrict__ Y, int C) {
  __shared__ float s1[256], s2[256];
  int row = blockIdx.x;
  const float* x = X + (size_t)row * C;
  float a = 0.f, q = 0.f;
  for (int c = threadIdx.x; c < C; c += blockDim.x) {
    float v = x[c]; a += v; q += v * v;
  }
  s1[threadIdx.x] = a; s2[threadIdx.x] = q;
  __syncthreads();
  for (int st = 128; st > 0; st >>= 1) {
    if ((int)threadIdx.x < st) {
      s1[threadIdx.x] += s1[threadIdx.x + st];
      s2[threadIdx.x] += s2[threadIdx.x + st];
    }
    __syncthreads();
  }
  float mu = s1[0] / C;
  float var = s2[0] / C - mu * mu;
  float rs = rsqrtf(var + 1e-5f);
  for (int c = threadIdx.x; c < C; c += blockDim.x)
    Y[(size_t)row * C + c] = (x[c] - mu) * rs * g[c] + b[c];
}

// ---------------------------------------------------------------------------
// host driver
// ---------------------------------------------------------------------------
extern "C" void kernel_launch(void* const* d_in, const int* in_sizes, int n_in,
                              void* d_out, int out_size, void* d_ws, size_t ws_size,
                              hipStream_t stream) {
  const int N = 5000, E = 60000, G = 200;
  const int Ef = E + N;

  auto P = [&](int i) { return (const float*)d_in[i]; };
  const float* x          = P(0);
  const int*   edge_index = (const int*)d_in[1];
  const float* edge_attr  = P(2);
  const int*   batch      = (const int*)d_in[3];
  const float* conditions = P(4);
  const float* ne_w = P(5), *ne_b = P(6), *ne_g = P(7), *ne_beta = P(8);
  const float* ee_w = P(9), *ee_b = P(10);
  auto GAT = [&](int l, int k) { return P(11 + 9 * l + k); };
  // gat params: 0 wl, 1 bl, 2 wr, 3 br, 4 we, 5 att, 6 bias, 7 bn_g, 8 bn_b
  const float* s2s_wih = P(47), *s2s_whh = P(48), *s2s_bih = P(49), *s2s_bhh = P(50);
  const float* ci_gw = P(51), *ci_gb = P(52);
  const float* ci_c1w = P(53), *ci_c1b = P(54), *ci_c1g = P(55), *ci_c1beta = P(56);
  const float* ci_c2w = P(57), *ci_c2b = P(58);
  const float* ci_wv = P(63), *ci_bv = P(64), *ci_wo = P(65), *ci_bo = P(66);
  const float* ci_ln_g = P(67), *ci_ln_b = P(68);
  const float* p1w = P(69), *p1b = P(70), *p1g = P(71), *p1beta = P(72);
  const float* p2w = P(73), *p2b = P(74), *p2g = P(75), *p2beta = P(76);
  const float* p3w = P(77), *p3b = P(78);

  // bump allocator in d_ws (~200 MB total)
  char* base = (char*)d_ws;
  size_t off = 0;
  auto alloc = [&](size_t bytes) -> void* {
    off = (off + 255) & ~(size_t)255;
    void* p = base + off;
    off += bytes;
    return p;
  };
  unsigned short* x_bf   = (unsigned short*)alloc((size_t)N * 64 * 2);
  unsigned* neww_sw      = (unsigned*)alloc((size_t)64 * 256 * 2);
  float* h    = (float*)alloc((size_t)N * 256 * 4);
  float* t1   = (float*)alloc((size_t)N * 256 * 4);
  float* t2   = (float*)alloc((size_t)N * 256 * 4);
  float* hres = (float*)alloc((size_t)N * 256 * 4);
  unsigned short* hin_bf = (unsigned short*)alloc((size_t)N * 256 * 2);
  float* e_f  = (float*)alloc((size_t)Ef * 256 * 4);
  unsigned short* ef_bf = (unsigned short*)alloc((size_t)Ef * 256 * 2);
  float* degb = (float*)alloc((size_t)N * 4);
  unsigned* wsw = (unsigned*)alloc((size_t)12 * 256 * 1024 * 2);  // 12 swizzled mats
  float* xl   = (float*)alloc((size_t)N * 1024 * 4);
  float* xr   = (float*)alloc((size_t)N * 1024 * 4);
  float* gout = (float*)alloc((size_t)N * 1024 * 4);
  float* logit= (float*)alloc((size_t)Ef * 4 * 4);
  float* exb  = (float*)alloc((size_t)Ef * 4 * 4);
  unsigned* segm = (unsigned*)alloc((size_t)N * 4 * 4);
  float* segden  = (float*)alloc((size_t)N * 4 * 4);
  float* ybuf = (float*)alloc((size_t)N * 256 * 4);
  float* meanb= (float*)alloc(512 * 4);
  float* varb = (float*)alloc(512 * 4);
  int* src_f  = (int*)alloc((size_t)Ef * 4);
  int* dst_f  = (int*)alloc((size_t)Ef * 4);
  float* qstar= (float*)alloc((size_t)G * 512 * 4);
  float* hlb  = (float*)alloc((size_t)G * 256 * 4);
  float* clb  = (float*)alloc((size_t)G * 256 * 4);
  float* gates= (float*)alloc((size_t)G * 1024 * 4);
  float* en   = (float*)alloc((size_t)N * 4);
  unsigned* gm= (unsigned*)alloc((size_t)G * 4);
  float* gden = (float*)alloc((size_t)G * 4);
  float* ea   = (float*)alloc((size_t)N * 4);
  float* rbuf = (float*)alloc((size_t)G * 256 * 4);
  float* gp   = (float*)alloc((size_t)G * 256 * 4);
  float* c1z  = (float*)alloc((size_t)G * 512 * 4);
  float* ceb  = (float*)alloc((size_t)G * 256 * 4);
  float* vb   = (float*)alloc((size_t)G * 256 * 4);
  float* u1   = (float*)alloc((size_t)G * 128 * 4);
  float* u2   = (float*)alloc((size_t)G * 64 * 4);
  (void)ws_size; (void)n_in; (void)in_sizes; (void)out_size;

  auto gb = [](long long n) { return dim3((unsigned)((n + 255) / 256)); };
  const unsigned WSLICE_DW = 256 * 1024 / 2;   // dwords per swizzled 256x1024 mat
  auto wl_sw = [&](int l) { return wsw + (size_t)(l * 3 + 0) * WSLICE_DW; };
  auto wr_sw = [&](int l) { return wsw + (size_t)(l * 3 + 1) * WSLICE_DW; };
  auto we_sw = [&](int l) { return wsw + (size_t)(l * 3 + 2) * WSLICE_DW; };

  // ---- input conversion + weight swizzles ----
  k_f32_to_bf16<<<gb((long long)N * 64), 256, 0, stream>>>(x, x_bf, (long long)N * 64);
  k_swizzle_w<<<gb(64 * 256 / 2), 256, 0, stream>>>(ne_w, neww_sw, 64, 256);
  for (int l = 0; l < 4; l++) {
    k_swizzle_w<<<gb(WSLICE_DW), 256, 0, stream>>>(GAT(l, 0), wl_sw(l), 256, 1024);
    k_swizzle_w<<<gb(WSLICE_DW), 256, 0, stream>>>(GAT(l, 2), wr_sw(l), 256, 1024);
    k_swizzle_w<<<gb(WSLICE_DW), 256, 0, stream>>>(GAT(l, 4), we_sw(l), 256, 1024);
  }

  // ---- node encoder: h = relu(bn(x @ ne_w + ne_b)) ----
  k_wmma_gemm_bias<<<dim3((N + 63) / 64, 256 / 64), 128, 0, stream>>>(
      x_bf, neww_sw, ne_b, t1, N, 256, 64);
  k_colstats<<<256, 256, 0, stream>>>(t1, meanb, varb, N, 256);
  k_bn_apply<<<gb((long long)N * 256), 256, 0, stream>>>(
      t1, meanb, varb, ne_g, ne_beta, nullptr, h, N, 256, 1);

  // ---- edge encoder + self loops ----
  k_sgemm<<<gb((long long)E * 256), 256, 0, stream>>>(edge_attr, ee_w, ee_b, e_f,
                                                      E, 256, 6, 0);
  k_build_edges<<<gb(Ef), 256, 0, stream>>>(edge_index, src_f, dst_f, E, N);
  hipMemsetAsync(degb, 0, (size_t)N * 4, stream);
  hipMemsetAsync(e_f + (size_t)E * 256, 0, (size_t)N * 256 * 4, stream);
  k_deg<<<gb(E), 256, 0, stream>>>(dst_f, degb, E);
  k_loop_sum<<<gb((long long)E * 256), 256, 0, stream>>>(e_f, dst_f,
                                                         e_f + (size_t)E * 256, E);
  k_loop_div<<<gb((long long)N * 256), 256, 0, stream>>>(e_f + (size_t)E * 256,
                                                         degb, N);
  k_f32_to_bf16<<<gb((long long)Ef * 256), 256, 0, stream>>>(e_f, ef_bf,
                                                             (long long)Ef * 256);

  // ---- 4 GAT layers ----
  const float* hcur = h;
  float* houts[4] = {t1, t2, t1, hres};
  for (int l = 0; l < 4; l++) {
    k_f32_to_bf16<<<gb((long long)N * 256), 256, 0, stream>>>(hcur, hin_bf,
                                                              (long long)N * 256);
    k_wmma_gemm_bias<<<dim3((N + 63) / 64, 1024 / 64), 128, 0, stream>>>(
        hin_bf, wl_sw(l), GAT(l, 1), xl, N, 1024, 256);
    k_wmma_gemm_bias<<<dim3((N + 63) / 64, 1024 / 64), 128, 0, stream>>>(
        hin_bf, wr_sw(l), GAT(l, 3), xr, N, 1024, 256);
    k_gat_edge_logits<<<(Ef + 63) / 64, 128, 0, stream>>>(
        ef_bf, we_sw(l), xl, xr, src_f, dst_f, GAT(l, 5), logit, Ef);
    k_fill_u32<<<gb((long long)N * 4), 256, 0, stream>>>(segm, NEG_INF_ORD,
                                                         (long long)N * 4);
    hipMemsetAsync(segden, 0, (size_t)N * 4 * 4, stream);
    k_edge_seg_max<<<gb((long long)Ef * 4), 256, 0, stream>>>(logit, dst_f, segm, Ef);
    k_edge_exp<<<gb((long long)Ef * 4), 256, 0, stream>>>(logit, dst_f, segm, exb,
                                                          segden, Ef);
    hipMemsetAsync(gout, 0, (size_t)N * 1024 * 4, stream);
    k_gat_scatter<<<gb((long long)Ef * 1024), 256, 0, stream>>>(
        xl, src_f, dst_f, exb, segden, gout, (long long)Ef * 1024);
    k_head_mean<<<gb((long long)N * 256), 256, 0, stream>>>(gout, GAT(l, 6), ybuf, N);
    k_colstats<<<256, 256, 0, stream>>>(ybuf, meanb, varb, N, 256);
    // layers 0-2: elu; layer 3: no activation + residual (+h)
    k_bn_apply<<<gb((long long)N * 256), 256, 0, stream>>>(
        ybuf, meanb, varb, GAT(l, 7), GAT(l, 8), (l == 3) ? h : nullptr, houts[l],
        N, 256, (l == 3) ? 0 : 2);
    hcur = houts[l];
  }
  const float* hres_p = houts[3];

  // ---- Set2Set (3 steps) ----
  hipMemsetAsync(qstar, 0, (size_t)G * 512 * 4, stream);
  hipMemsetAsync(hlb, 0, (size_t)G * 256 * 4, stream);
  hipMemsetAsync(clb, 0, (size_t)G * 256 * 4, stream);
  for (int it = 0; it < 3; it++) {
    k_sgemm<<<gb((long long)G * 1024), 256, 0, stream>>>(qstar, s2s_wih, s2s_bih,
                                                         gates, G, 1024, 512, 0);
    k_sgemm<<<gb((long long)G * 1024), 256, 0, stream>>>(hlb, s2s_whh, s2s_bhh,
                                                         gates, G, 1024, 256, 1);
    k_lstm_update<<<gb((long long)G * 256), 256, 0, stream>>>(gates, hlb, clb, G);
    k_s2s_en<<<gb(N), 256, 0, stream>>>(hres_p, hlb, batch, en, N);
    k_fill_u32<<<gb(G), 256, 0, stream>>>(gm, NEG_INF_ORD, G);
    hipMemsetAsync(gden, 0, (size_t)G * 4, stream);
    k_seg_max_n<<<gb(N), 256, 0, stream>>>(en, batch, gm, N);
    k_seg_exp_n<<<gb(N), 256, 0, stream>>>(en, batch, gm, ea, gden, N);
    hipMemsetAsync(rbuf, 0, (size_t)G * 256 * 4, stream);
    k_s2s_r<<<gb((long long)N * 256), 256, 0, stream>>>(hres_p, batch, ea, gden,
                                                        rbuf, N);
    k_qstar<<<gb((long long)G * 512), 256, 0, stream>>>(hlb, rbuf, qstar, G);
  }

  // ---- condition integration ----
  k_sgemm<<<gb((long long)G * 256), 256, 0, stream>>>(qstar, ci_gw, ci_gb, gp,
                                                      G, 256, 512, 0);
  k_sgemm<<<gb((long long)G * 512), 256, 0, stream>>>(conditions, ci_c1w, ci_c1b,
                                                      c1z, G, 512, 32, 0);
  k_colstats<<<512, 256, 0, stream>>>(c1z, meanb, varb, G, 512);
  k_bn_apply<<<gb((long long)G * 512), 256, 0, stream>>>(
      c1z, meanb, varb, ci_c1g, ci_c1beta, nullptr, c1z, G, 512, 1);
  k_sgemm<<<gb((long long)G * 256), 256, 0, stream>>>(c1z, ci_c2w, ci_c2b, ceb,
                                                      G, 256, 512, 0);
  // MHA with seq_len 1 reduces to out_proj(v_proj(ce))
  k_sgemm<<<gb((long long)G * 256), 256, 0, stream>>>(ceb, ci_wv, ci_bv, vb,
                                                      G, 256, 256, 0);
  k_sgemm<<<gb((long long)G * 256), 256, 0, stream>>>(vb, ci_wo, ci_bo, gp,
                                                      G, 256, 256, 1);  // gp += att
  k_layernorm<<<G, 256, 0, stream>>>(gp, ci_ln_g, ci_ln_b, gp, 256);

  // ---- predictor MLP ----
  k_sgemm<<<gb((long long)G * 128), 256, 0, stream>>>(gp, p1w, p1b, u1, G, 128, 256, 0);
  k_colstats<<<128, 256, 0, stream>>>(u1, meanb, varb, G, 128);
  k_bn_apply<<<gb((long long)G * 128), 256, 0, stream>>>(u1, meanb, varb, p1g,
                                                         p1beta, nullptr, u1, G, 128, 1);
  k_sgemm<<<gb((long long)G * 64), 256, 0, stream>>>(u1, p2w, p2b, u2, G, 64, 128, 0);
  k_colstats<<<64, 256, 0, stream>>>(u2, meanb, varb, G, 64);
  k_bn_apply<<<gb((long long)G * 64), 256, 0, stream>>>(u2, meanb, varb, p2g,
                                                        p2beta, nullptr, u2, G, 64, 1);
  k_sgemm<<<gb(G), 256, 0, stream>>>(u2, p3w, p3b, (float*)d_out, G, 1, 64, 0);
}
